// PointCloudEncoder_84877143704307
// MI455X (gfx1250) — matrix operand order repaired and checked by reading, not compiled, not run on hardware
//
#include <hip/hip_runtime.h>
#include <hip/hip_bf16.h>

typedef __attribute__((ext_vector_type(16))) _Float16 v16h;
typedef __attribute__((ext_vector_type(8)))  float    v8f;

#define BB   128
#define NN   512
#define KK   10
#define BPP  8
#define BKK  4

// ---------------- WMMA helper (CDNA5 wave32, 16x16x32 f16 -> f32) ----------------

__device__ __forceinline__ v8f wmma_f32_f16(v16h a, v16h b, v8f c) {
  return __builtin_amdgcn_wmma_f32_16x16x32_f16(
      /*neg_a=*/false, a, /*neg_b=*/false, b,
      /*c_mod=*/(short)0, c, /*reuse_a=*/false, /*reuse_b=*/false);
}

// Fragment-layout index maps (contiguous k = 16h + e order used here, wave32, 16-bit A 16x32 / B 32x16):
// A element (m, k): lane = m + 16*((k>>4)&1), e = k & 15
// B element (k, c): lane = c + 16*(k>>4),     e = k & 15
__device__ __forceinline__ int amap_lane(int m, int k) { return m + 16 * ((k >> 4) & 1); }
__device__ __forceinline__ int amap_e(int k)           { return k & 15; }
__device__ __forceinline__ int bmap_lane(int c, int k) { return (c & 15) + 16 * (k >> 4); }
__device__ __forceinline__ int bmap_e(int k)           { return k & 15; }

__device__ __forceinline__ void atomicMaxF(float* addr, float v) {
  // valid for non-negative floats (relu6 outputs), buffer pre-zeroed
  atomicMax((int*)addr, __float_as_int(v));
}

__device__ __forceinline__ float relu6f(float x) { return fminf(fmaxf(x, 0.f), 6.f); }

// Async copy of 16 contiguous rows (16*C*4 bytes, 16B-aligned) into LDS.
// Issues (16*C*4)/512 GLOBAL_LOAD_ASYNC_TO_LDS_B128 ops (tracked by ASYNCcnt).
template <int C>
__device__ __forceinline__ void async_stage16(const float* gbase, unsigned ldsbase, int lane) {
  constexpr int QISS = (16 * C * 4) / 512;  // 32 lanes * 16B per issue
#pragma unroll
  for (int q = 0; q < QISS; ++q) {
    unsigned laddr = ldsbase + (unsigned)(q * 512 + lane * 16);
    const float* ga = gbase + (q * 512 + lane * 16) / 4;
    asm volatile("global_load_async_to_lds_b128 %0, %1, off"
                 :: "v"(laddr), "v"(ga)
                 : "memory");
  }
}

// ---------------- squared norms per point ----------------
__global__ void k_sqnorm(const float* __restrict__ x, float* __restrict__ sqn,
                         int npts, int C) {
  int i = blockIdx.x * blockDim.x + threadIdx.x;
  if (i >= npts) return;
  float s = 0.f;
  for (int c = 0; c < C; ++c) { float v = x[(size_t)i * C + c]; s += v * v; }
  sqn[i] = s;
}

// ---------------- f32 weights -> f16, pre-swizzled to B-fragment layout ----------------
// Output layout: wh[((kc*4 + ct)*32 + lane)*16 + e], chunks kc of 32 k-rows,
// col tiles ct of 16 columns.  Rows kk >= C2 are zero padding.
__global__ void k_convw_frag(const float* __restrict__ w, _Float16* __restrict__ wh,
                             int C2, int KCH) {
  int o = blockIdx.x * blockDim.x + threadIdx.x;
  int total = KCH * 2048;            // KCH * 4ct * 32lane * 16e
  if (o >= total) return;
  int kc   = o >> 11;
  int r    = o & 2047;
  int ct   = r >> 9;
  int lane = (r >> 4) & 31;
  int e    = r & 15;
  int col  = ct * 16 + (lane & 15);
  int kloc = ((lane >> 4) << 4) | e;
  int kk   = kc * 32 + kloc;
  wh[o] = (kk < C2) ? (_Float16)w[kk * 64 + col] : (_Float16)0.f;
}

// ---------------- KNN via WMMA gram matrix ----------------
// grid: B * (N/16) blocks, 32 threads (1 wave). One 16-row tile per wave.
// For C=64 the column tiles are double-buffered into LDS with async loads
// overlapped against the WMMA + top-k of the previous tile.
template <int C>
__global__ void k_knn(const float* __restrict__ xin, const float* __restrict__ sqn,
                      int* __restrict__ idx) {
  constexpr int KCH = (C + 31) / 32;
  constexpr bool ASYNC = (C % 32 == 0);
  const int lane = threadIdx.x;
  const int b  = blockIdx.x >> 5;        // 32 row-tiles per batch
  const int i0 = (blockIdx.x & 31) << 4;

  __shared__ __align__(32) _Float16 Afrag[KCH][32][16];
  __shared__ __align__(32) _Float16 Bfrag[KCH][32][16];
  __shared__ __align__(16) float Xstage[ASYNC ? 2 * 16 * C : 1];
  __shared__ float dls[16 * 16];
  __shared__ float sqi[16];

  // zero (covers channel padding when C % 32 != 0)
  if (C % 32 != 0) {
    for (int i = lane; i < KCH * 512; i += 32)
      (&Afrag[0][0][0])[i] = (_Float16)0.f;
    for (int i = lane; i < KCH * 512; i += 32)
      (&Bfrag[0][0][0])[i] = (_Float16)0.f;
    __syncthreads();
  }
  // stage A tile in fragment order
  for (int i = lane; i < 16 * C; i += 32) {
    int m = i / C, c = i % C;
    int kc = c >> 5, kl = c & 31;
    Afrag[kc][amap_lane(m, kl)][amap_e(kl)] =
        (_Float16)xin[(size_t)(b * NN + i0 + m) * C + c];
  }
  if (lane < 16) sqi[lane] = sqn[b * NN + i0 + lane];
  __syncthreads();

  v16h afr[KCH];
#pragma unroll
  for (int kc = 0; kc < KCH; ++kc)
    afr[kc] = *(const v16h*)&Afrag[kc][lane][0];

  float bd[KK]; int bi[KK];
#pragma unroll
  for (int t = 0; t < KK; ++t) { bd[t] = 3.0e38f; bi[t] = 0; }

  const int g = lane >> 4, cx = lane & 15;

  unsigned xsbase = 0;
  if constexpr (ASYNC) {
    xsbase = (unsigned)(size_t)&Xstage[0];             // LDS address from low 32 bits
    async_stage16<C>(&xin[(size_t)(b * NN) * C], xsbase, lane);   // tile j0=0 -> buf 0
  }

  for (int j0 = 0; j0 < NN; j0 += 16) {
    __syncthreads();
    if constexpr (ASYNC) {
      const int cur = (j0 >> 4) & 1;
      if (j0 + 16 < NN) {
        async_stage16<C>(&xin[(size_t)(b * NN + j0 + 16) * C],
                         xsbase + (unsigned)((cur ^ 1) * 16 * C * 4), lane);
        asm volatile("s_wait_asynccnt %0" :: "i"((16 * C * 4) / 512) : "memory");
      } else {
        asm volatile("s_wait_asynccnt 0" ::: "memory");
      }
      const float* xs = &Xstage[cur * 16 * C];
      for (int i = lane; i < 16 * C; i += 32) {
        int jj = i / C, c = i % C;
        int kc = c >> 5, kl = c & 31;
        Bfrag[kc][bmap_lane(jj, kl)][bmap_e(kl)] = (_Float16)xs[i];
      }
    } else {
      // direct path (C==3: 12B row stride, not 16B-aligned for async B128)
      for (int i = lane; i < 16 * C; i += 32) {
        int jj = i / C, c = i % C;
        int kc = c >> 5, kl = c & 31;
        Bfrag[kc][bmap_lane(jj, kl)][bmap_e(kl)] =
            (_Float16)xin[(size_t)(b * NN + j0 + jj) * C + c];
      }
      if (j0 + 16 < NN)
        __builtin_prefetch(&xin[(size_t)(b * NN + j0 + 16) * C], 0, 1);
    }
    __syncthreads();

    v8f acc = {0.f, 0.f, 0.f, 0.f, 0.f, 0.f, 0.f, 0.f};
#pragma unroll
    for (int kc = 0; kc < KCH; ++kc) {
      v16h bfr = *(const v16h*)&Bfrag[kc][lane][0];
      acc = wmma_f32_f16(afr[kc], bfr, acc);
    }

    float sqj = sqn[b * NN + j0 + cx];
#pragma unroll
    for (int v = 0; v < 8; ++v) {
      int row = v + 8 * g;
      dls[row * 16 + cx] = sqi[row] - 2.f * acc[v] + sqj;
    }
    __syncthreads();

    if (lane < 16) {
#pragma unroll
      for (int c16 = 0; c16 < 16; ++c16) {
        float d = dls[lane * 16 + c16];
        int   id = j0 + c16;
        if (d < bd[KK - 1]) {
          bd[KK - 1] = d; bi[KK - 1] = id;
#pragma unroll
          for (int t = KK - 1; t > 0; --t) {
            if (bd[t] < bd[t - 1]) {
              float td = bd[t]; bd[t] = bd[t - 1]; bd[t - 1] = td;
              int   ti = bi[t]; bi[t] = bi[t - 1]; bi[t - 1] = ti;
            }
          }
        }
      }
    }
  }
  __syncthreads();
  if (lane < 16) {
#pragma unroll
    for (int t = 0; t < KK; ++t)
      idx[(size_t)(b * NN + i0 + lane) * KK + t] = bi[t];
  }
}

// ---------------- Edge conv (WMMA) ----------------
// grid: B*N blocks, 32 threads. MODE: 0=BN stats, 1=BN apply+relu6+maxK, 2=relu6+maxK.
template <int Cin, int MODE>
__global__ void k_edgeconv(const float* __restrict__ xin, const int* __restrict__ idx,
                           const _Float16* __restrict__ whfrag,
                           const float* __restrict__ bias,
                           const float* __restrict__ bnmean, const float* __restrict__ bnrstd,
                           const float* __restrict__ gamma, const float* __restrict__ beta,
                           float* __restrict__ fout, float* __restrict__ fmx,
                           float* __restrict__ bnsum, float* __restrict__ bnsq) {
  constexpr int C2  = 2 * Cin;
  constexpr int KCH = (C2 + 31) / 32;
  const int lane = threadIdx.x;
  const int b = blockIdx.x >> 9;
  const int n = blockIdx.x & (NN - 1);

  __shared__ __align__(32) _Float16 Afrag[KCH][32][16];
  __shared__ float ht[16 * 64];
  __shared__ float lsum[64], lsq[64];

  for (int i = lane; i < KCH * 512; i += 32)
    (&Afrag[0][0][0])[i] = (_Float16)0.f;   // row (k>=10) + channel padding
  if (MODE == 0) {
    for (int i = lane; i < 64; i += 32) { lsum[i] = 0.f; lsq[i] = 0.f; }
  }
  __syncthreads();

  // build edge features directly in A-fragment order
  const float* ctr = &xin[(size_t)(b * NN + n) * Cin];
  for (int i = lane; i < KK * Cin; i += 32) {
    int k = i / Cin, c = i % Cin;
    int j = idx[(size_t)(b * NN + n) * KK + k];
    float cc = ctr[c];
    float nb = xin[(size_t)(b * NN + j) * Cin + c];
    int c0 = c, c1 = Cin + c;    // [nbr - ctr | ctr]
    Afrag[c0 >> 5][amap_lane(k, c0 & 31)][amap_e(c0 & 31)] = (_Float16)(nb - cc);
    Afrag[c1 >> 5][amap_lane(k, c1 & 31)][amap_e(c1 & 31)] = (_Float16)cc;
  }
  __syncthreads();

  const v16h* wf = (const v16h*)whfrag;   // [(kc*4+ct)*32 + lane]
  v8f acc[4];
#pragma unroll
  for (int t = 0; t < 4; ++t) acc[t] = (v8f){0.f,0.f,0.f,0.f,0.f,0.f,0.f,0.f};

#pragma unroll
  for (int kc = 0; kc < KCH; ++kc) {
    v16h afr = *(const v16h*)&Afrag[kc][lane][0];
#pragma unroll
    for (int ct = 0; ct < 4; ++ct) {
      v16h bfr = wf[(kc * 4 + ct) * 32 + lane];
      acc[ct] = wmma_f32_f16(afr, bfr, acc[ct]);
    }
  }

  const int g = lane >> 4, cx = lane & 15;
#pragma unroll
  for (int ct = 0; ct < 4; ++ct) {
    const int col = ct * 16 + cx;
    const float bv = bias[col];
#pragma unroll
    for (int v = 0; v < 8; ++v) {
      int row = v + 8 * g;
      float hv = acc[ct][v] + bv;
      if (MODE == 0) {
        if (row < KK) { atomicAdd(&lsum[col], hv); atomicAdd(&lsq[col], hv * hv); }
      } else {
        if (MODE == 1)
          hv = (hv - bnmean[col]) * bnrstd[col] * gamma[col] + beta[col];
        hv = relu6f(hv);
        if (row < KK) ht[row * 64 + col] = hv;
      }
    }
  }
  __syncthreads();

  if (MODE == 0) {
    for (int i = lane; i < 64; i += 32) {
      atomicAdd(&bnsum[i], lsum[i]);
      atomicAdd(&bnsq[i],  lsq[i]);
    }
  } else {
    for (int col = lane; col < 64; col += 32) {
      float m = 0.f;
#pragma unroll
      for (int r = 0; r < KK; ++r) m = fmaxf(m, ht[r * 64 + col]);
      fout[(size_t)(b * NN + n) * 64 + col] = m;
      atomicMaxF(&fmx[b * 64 + col], m);
    }
  }
}

// ---------------- BN finalize ----------------
__global__ void k_bnfin(const float* __restrict__ s, const float* __restrict__ ss,
                        float invcnt, float* __restrict__ mean, float* __restrict__ rstd) {
  int c = threadIdx.x;  // 64
  float m = s[c] * invcnt;
  float v = ss[c] * invcnt - m * m;
  mean[c] = m;
  rstd[c] = rsqrtf(v + 1e-5f);
}

// ---------------- bbox KNN (8 points, k=4) ----------------
__global__ void k_bbox_knn(const float* __restrict__ x, int* __restrict__ idxbb, int C) {
  int b = blockIdx.x, t = threadIdx.x;
  __shared__ float xl[BPP * 64];
  for (int i = t; i < BPP * C; i += 32) xl[i] = x[(size_t)b * BPP * C + i];
  __syncthreads();
  if (t < BPP) {
    float d[BPP];
    for (int j = 0; j < BPP; ++j) {
      float s = 0.f;
      for (int c = 0; c < C; ++c) { float df = xl[t * C + c] - xl[j * C + c]; s += df * df; }
      d[j] = s;
    }
    for (int sel = 0; sel < BKK; ++sel) {
      int bj = 0; float bv = 3.0e38f;
      for (int j = 0; j < BPP; ++j) if (d[j] < bv) { bv = d[j]; bj = j; }
      idxbb[(b * BPP + t) * BKK + sel] = bj;
      d[bj] = 3.0e38f;
    }
  }
}

// ---------------- bbox edge conv (scalar; tiny) ----------------
// grid: B*BP*BK blocks, 64 threads (one per out channel). mode: 0=stats, 1=apply.
__global__ void k_bbox_edge(const float* __restrict__ xin, const int* __restrict__ idxbb,
                            const float* __restrict__ w, const float* __restrict__ bias,
                            int Cin, int mode,
                            const float* __restrict__ bnmean, const float* __restrict__ bnrstd,
                            const float* __restrict__ gamma, const float* __restrict__ beta,
                            float* __restrict__ gout,
                            float* __restrict__ bnsum, float* __restrict__ bnsq) {
  int t = threadIdx.x;
  int blk = blockIdx.x;
  int k = blk & 3, p = (blk >> 2) & 7, b = blk >> 5;
  __shared__ float ctr[64], nbr[64];
  int j = idxbb[(b * BPP + p) * BKK + k];
  if (t < Cin) {
    ctr[t] = xin[(size_t)(b * BPP + p) * Cin + t];
    nbr[t] = xin[(size_t)(b * BPP + j) * Cin + t];
  }
  __syncthreads();
  float h = bias[t];
  for (int c = 0; c < Cin; ++c) h += (nbr[c] - ctr[c]) * w[c * 64 + t];
  for (int c = 0; c < Cin; ++c) h += ctr[c] * w[(Cin + c) * 64 + t];
  if (mode == 0) {
    atomicAdd(&bnsum[t], h);
    atomicAdd(&bnsq[t], h * h);
  } else {
    h = (h - bnmean[t]) * bnrstd[t] * gamma[t] + beta[t];
    h = relu6f(h);
    atomicMaxF(&gout[(size_t)(b * BPP + p) * 64 + t], h);
  }
}

// ---------------- g = concat(g1,g2).max(axis=1) ----------------
__global__ void k_gmax(const float* __restrict__ g1, const float* __restrict__ g2,
                       float* __restrict__ gmx) {
  int b = blockIdx.x, t = threadIdx.x;  // 128 threads
  const float* src = (t < 64) ? g1 : g2;
  int c = t & 63;
  float m = 0.f;
  for (int p = 0; p < BPP; ++p) m = fmaxf(m, src[(size_t)(b * BPP + p) * 64 + c]);
  gmx[b * 128 + t] = m;
}

// ---------------- fb = relu6(g @ Wp + b) ----------------
__global__ void k_fb(const float* __restrict__ gmx, const float* __restrict__ w,
                     const float* __restrict__ bias, float* __restrict__ fb) {
  int b = blockIdx.x, t = threadIdx.x;  // 64 threads
  float s = bias[t];
  for (int i = 0; i < 128; ++i) s += gmx[b * 128 + i] * w[i * 64 + t];
  fb[b * 64 + t] = relu6f(s);
}

// ---------------- final: tanh(proj) + L2 normalize ----------------
__global__ void k_final(const float* __restrict__ fmaxs, const float* __restrict__ fb,
                        const float* __restrict__ pw, const float* __restrict__ pb,
                        float* __restrict__ out) {
  int b = blockIdx.x, t = threadIdx.x;  // 128 threads
  __shared__ float fv[320];
  __shared__ float red[128];
  for (int i = t; i < 320; i += 128)
    fv[i] = (i < 256) ? fmaxs[(size_t)(i >> 6) * (BB * 64) + b * 64 + (i & 63)]
                      : fb[b * 64 + (i - 256)];
  __syncthreads();
  float s = pb[t];
  for (int i = 0; i < 320; ++i) s += fv[i] * pw[i * 128 + t];
  float tv = tanhf(s);
  red[t] = tv * tv;
  __syncthreads();
  for (int off = 64; off > 0; off >>= 1) {
    if (t < off) red[t] += red[t + off];
    __syncthreads();
  }
  float nrm = sqrtf(red[0]);
  out[b * 128 + t] = tv / (nrm + 1e-9f);
}

// ---------------- host orchestration ----------------
extern "C" void kernel_launch(void* const* d_in, const int* in_sizes, int n_in,
                              void* d_out, int out_size, void* d_ws, size_t ws_size,
                              hipStream_t stream) {
  (void)in_sizes; (void)n_in; (void)out_size; (void)ws_size;

  const float* x    = (const float*)d_in[0];
  const float* bbox = (const float*)d_in[2];
  const float* gc_w[4]  = {(const float*)d_in[3], (const float*)d_in[5],
                           (const float*)d_in[7], (const float*)d_in[9]};
  const float* gc_b[4]  = {(const float*)d_in[4], (const float*)d_in[6],
                           (const float*)d_in[8], (const float*)d_in[10]};
  const float* gc_gm[4] = {(const float*)d_in[11], nullptr, (const float*)d_in[13], nullptr};
  const float* gc_bt[4] = {(const float*)d_in[12], nullptr, (const float*)d_in[14], nullptr};
  const float* bb1_w = (const float*)d_in[15]; const float* bb1_b = (const float*)d_in[16];
  const float* bb1_g = (const float*)d_in[17]; const float* bb1_be = (const float*)d_in[18];
  const float* bb2_w = (const float*)d_in[19]; const float* bb2_b = (const float*)d_in[20];
  const float* bb2_g = (const float*)d_in[21]; const float* bb2_be = (const float*)d_in[22];
  const float* bbp_w = (const float*)d_in[23]; const float* bbp_b = (const float*)d_in[24];
  const float* pr_w  = (const float*)d_in[25]; const float* pr_b  = (const float*)d_in[26];

  // workspace carve-up
  char* ws = (char*)d_ws;
  size_t off = 0;
  auto carve = [&](size_t bytes) -> void* {
    void* p = ws + off;
    off = (off + bytes + 255) & ~(size_t)255;
    return p;
  };
  float*    sqn   = (float*)carve((size_t)BB * NN * 4);
  int*      idx   = (int*)carve((size_t)BB * NN * KK * 4);
  float*    fA    = (float*)carve((size_t)BB * NN * 64 * 4);
  float*    fBf   = (float*)carve((size_t)BB * NN * 64 * 4);
  float*    fmaxs = (float*)carve((size_t)4 * BB * 64 * 4);
  _Float16* wh    = (_Float16*)carve((size_t)4 * 2048 * 2);
  float*    bnsum = (float*)carve(64 * 4);
  float*    bnsq  = (float*)carve(64 * 4);
  float*    bnmean= (float*)carve(64 * 4);
  float*    bnrstd= (float*)carve(64 * 4);
  int*      idxbb = (int*)carve((size_t)BB * BPP * BKK * 4);
  float*    g1    = (float*)carve((size_t)BB * BPP * 64 * 4);
  float*    g2    = (float*)carve((size_t)BB * BPP * 64 * 4);
  float*    gmx   = (float*)carve((size_t)BB * 128 * 4);
  float*    fb    = (float*)carve((size_t)BB * 64 * 4);

  hipMemsetAsync(fmaxs, 0, (size_t)4 * BB * 64 * 4, stream);
  hipMemsetAsync(g1, 0, (size_t)BB * BPP * 64 * 4, stream);
  hipMemsetAsync(g2, 0, (size_t)BB * BPP * 64 * 4, stream);

  // -------- point branch: 4 graph-conv layers --------
  const float* fprev = x;
  float* fbufs[2] = {fA, fBf};
  for (int l = 0; l < 4; ++l) {
    const int  C  = (l == 0) ? 3 : 64;
    const bool bn = (l == 0 || l == 2);
    float* fcur = fbufs[l & 1];
    float* fmxl = fmaxs + (size_t)l * BB * 64;
    const int C2 = 2 * C, KCH = (C2 + 31) / 32;

    k_sqnorm<<<(BB * NN + 255) / 256, 256, 0, stream>>>(fprev, sqn, BB * NN, C);
    if (C == 3) k_knn<3><<<BB * (NN / 16), 32, 0, stream>>>(fprev, sqn, idx);
    else        k_knn<64><<<BB * (NN / 16), 32, 0, stream>>>(fprev, sqn, idx);
    k_convw_frag<<<(KCH * 2048 + 255) / 256, 256, 0, stream>>>(gc_w[l], wh, C2, KCH);

    if (bn) {
      hipMemsetAsync(bnsum, 0, 64 * 4, stream);
      hipMemsetAsync(bnsq,  0, 64 * 4, stream);
      if (C == 3)
        k_edgeconv<3, 0><<<BB * NN, 32, 0, stream>>>(fprev, idx, wh, gc_b[l],
            nullptr, nullptr, nullptr, nullptr, nullptr, nullptr, bnsum, bnsq);
      else
        k_edgeconv<64, 0><<<BB * NN, 32, 0, stream>>>(fprev, idx, wh, gc_b[l],
            nullptr, nullptr, nullptr, nullptr, nullptr, nullptr, bnsum, bnsq);
      k_bnfin<<<1, 64, 0, stream>>>(bnsum, bnsq, 1.f / ((float)BB * NN * KK), bnmean, bnrstd);
      if (C == 3)
        k_edgeconv<3, 1><<<BB * NN, 32, 0, stream>>>(fprev, idx, wh, gc_b[l],
            bnmean, bnrstd, gc_gm[l], gc_bt[l], fcur, fmxl, nullptr, nullptr);
      else
        k_edgeconv<64, 1><<<BB * NN, 32, 0, stream>>>(fprev, idx, wh, gc_b[l],
            bnmean, bnrstd, gc_gm[l], gc_bt[l], fcur, fmxl, nullptr, nullptr);
    } else {
      k_edgeconv<64, 2><<<BB * NN, 32, 0, stream>>>(fprev, idx, wh, gc_b[l],
          nullptr, nullptr, nullptr, nullptr, fcur, fmxl, nullptr, nullptr);
    }
    fprev = fcur;
  }

  // -------- bbox branch --------
  k_bbox_knn<<<BB, 32, 0, stream>>>(bbox, idxbb, 3);
  hipMemsetAsync(bnsum, 0, 64 * 4, stream);
  hipMemsetAsync(bnsq,  0, 64 * 4, stream);
  k_bbox_edge<<<BB * BPP * BKK, 64, 0, stream>>>(bbox, idxbb, bb1_w, bb1_b, 3, 0,
      nullptr, nullptr, nullptr, nullptr, nullptr, bnsum, bnsq);
  k_bnfin<<<1, 64, 0, stream>>>(bnsum, bnsq, 1.f / ((float)BB * BPP * BKK), bnmean, bnrstd);
  k_bbox_edge<<<BB * BPP * BKK, 64, 0, stream>>>(bbox, idxbb, bb1_w, bb1_b, 3, 1,
      bnmean, bnrstd, bb1_g, bb1_be, g1, nullptr, nullptr);

  k_bbox_knn<<<BB, 32, 0, stream>>>(g1, idxbb, 64);
  hipMemsetAsync(bnsum, 0, 64 * 4, stream);
  hipMemsetAsync(bnsq,  0, 64 * 4, stream);
  k_bbox_edge<<<BB * BPP * BKK, 64, 0, stream>>>(g1, idxbb, bb2_w, bb2_b, 64, 0,
      nullptr, nullptr, nullptr, nullptr, nullptr, bnsum, bnsq);
  k_bnfin<<<1, 64, 0, stream>>>(bnsum, bnsq, 1.f / ((float)BB * BPP * BKK), bnmean, bnrstd);
  k_bbox_edge<<<BB * BPP * BKK, 64, 0, stream>>>(g1, idxbb, bb2_w, bb2_b, 64, 1,
      bnmean, bnrstd, bb2_g, bb2_be, g2, nullptr, nullptr);

  k_gmax<<<BB, 128, 0, stream>>>(g1, g2, gmx);
  k_fb<<<BB, 64, 0, stream>>>(gmx, bbp_w, bbp_b, fb);

  // -------- final projection + tanh + L2 normalize --------
  k_final<<<BB, 128, 0, stream>>>(fmaxs, fb, pr_w, pr_b, (float*)d_out);
}
